// DSA_5866925326622
// MI455X (gfx1250) — compile-verified, matrix operand order checked by reading, not run on hardware
//
#include <hip/hip_runtime.h>
#include <hip/hip_bf16.h>
#include <math.h>
#include <stdint.h>

#define Bc 8
#define Cc 256
#define Nc 4096
#define Kc 1024
#define NHc 8
#define HDc 32
#define CHc 64

typedef __bf16 bf16_t;
typedef __bf16 v16bf __attribute__((ext_vector_type(16)));
typedef float  v8f   __attribute__((ext_vector_type(8)));
typedef unsigned int u32x4 __attribute__((ext_vector_type(4)));
typedef int          i32x8 __attribute__((ext_vector_type(8)));
typedef int          i32x4 __attribute__((ext_vector_type(4)));

#if __has_include(<hip/amd_detail/amd_gfx1250_TDM.h>)
#define TDM_SIX_ARG 1
#else
#define TDM_SIX_ARG 0
#endif

// ---------------- WMMA helpers ----------------

__device__ inline v8f wmma_bf16(v16bf a, v16bf b, v8f c) {
  // D = A(16x32 bf16) x B(32x16 bf16) + C(16x16 f32)
  return __builtin_amdgcn_wmma_f32_16x16x32_bf16(false, a, false, b, (short)0, c, false, false);
}

// Load a 16x32 bf16 tile in WMMA A-layout (also valid as the B operand when the
// source matrix holds B^T rows, e.g. weight matrices stored [out, in]).
// Lane l: row = row0 + (l&15); elements K = col0 + (l>>4)*8 + {0..7} and +16.
__device__ inline v16bf load_tile(const bf16_t* __restrict__ base, int row0, int ld,
                                  int col0, int lane) {
  int r  = row0 + (lane & 15);
  int h8 = (lane >> 4) << 3;
  const bf16_t* p = base + (size_t)r * ld + col0 + h8;
  union { uint4 u[2]; v16bf v; } t;
  t.u[0] = *(const uint4*)(p);
  t.u[1] = *(const uint4*)(p + 16);
  return t.v;
}

// Same tile pattern but from a staged 16x256 LDS block (row stride 256).
__device__ inline v16bf load_tile_lds(const bf16_t* s, int col0, int lane) {
  int r  = lane & 15;
  int h8 = (lane >> 4) << 3;
  const bf16_t* p = s + r * Cc + col0 + h8;
  union { uint4 u[2]; v16bf v; } t;
  t.u[0] = *(const uint4*)(p);
  t.u[1] = *(const uint4*)(p + 16);
  return t.v;
}

// ---------------- Tensor Data Mover: stage 16 rows x 256 bf16 into LDS ----------
// D# per CDNA5 ISA ch.8: group0 = {flags/count, lds_addr, global_addr, type=2},
// group1 = {data_size=2B, tensor/tile dims, dim0 stride}.  2D tile, groups 2/3 zero.

__device__ inline void tdm_stage_16x256(bf16_t* s, const bf16_t* g) {
#if defined(__gfx1250__) && __has_builtin(__builtin_amdgcn_tensor_load_to_lds)
  unsigned long long ga = (unsigned long long)(uintptr_t)g;
  unsigned lds_off = (unsigned)(uintptr_t)s;        // low 32 bits = LDS byte offset
  u32x4 g0;
  g0[0] = 1u;                                       // count=1, user mode, no gather
  g0[1] = lds_off;                                  // lds_addr
  g0[2] = (unsigned)(ga & 0xFFFFFFFFu);             // global_addr[31:0]
  g0[3] = (unsigned)((ga >> 32) & 0x01FFFFFFu)      // global_addr[56:32]
        | (2u << 30);                               // type = 2 ("image")
  i32x8 g1;
  g1[0] = (int)(1u << 16);                          // data_size=1 -> 2 bytes; mask=0
  g1[1] = (int)((Cc & 0xFFFFu) << 16);              // tensor_dim0[15:0] = 256
  g1[2] = (int)((16u & 0xFFFFu) << 16);             // tensor_dim0[31:16]=0 | tensor_dim1[15:0]=16
  g1[3] = (int)((Cc & 0xFFFFu) << 16);              // tensor_dim1[31:16]=0 | tile_dim0=256
  g1[4] = 16;                                       // tile_dim1=16, tile_dim2=0
  g1[5] = Cc;                                       // tensor_dim0_stride[31:0] = 256
  g1[6] = 0;                                        // stride[47:32] | dim1_stride[15:0]
  g1[7] = 0;
  i32x4 z4 = {0, 0, 0, 0};
#if TDM_SIX_ARG
  i32x8 z8 = {0, 0, 0, 0, 0, 0, 0, 0};
  __builtin_amdgcn_tensor_load_to_lds(g0, g1, z4, z4, z8, 0);
#else
  __builtin_amdgcn_tensor_load_to_lds(g0, g1, z4, z4, 0);
#endif
  __builtin_amdgcn_s_wait_tensorcnt(0);
#else
  (void)s; (void)g;
#endif
}

#if defined(__gfx1250__) && __has_builtin(__builtin_amdgcn_tensor_load_to_lds)
#define HAVE_TDM 1
#else
#define HAVE_TDM 0
#endif

// ---------------- 1) importance net (exact fp32) ----------------

__global__ __launch_bounds__(256) void k_importance(
    const float* __restrict__ x, const float* __restrict__ bmap,
    const float* __restrict__ w1, const float* __restrict__ b1,
    const float* __restrict__ w2, const float* __restrict__ b2,
    float* __restrict__ imp_ws, float* __restrict__ imp_out)
{
  __shared__ float s_w1[CHc * 128];   // 32 KB chunk of W1
  const int tid = threadIdx.x;
  const int b = blockIdx.x >> 4;            // 16 blocks per batch
  const int p = ((blockIdx.x & 15) << 8) + tid;
  float acc[CHc];
  #pragma unroll
  for (int o = 0; o < CHc; ++o) acc[o] = b1[o];
  const float* xp = x + (size_t)b * Cc * Nc + p;
  for (int ch = 0; ch < 2; ++ch) {
    __syncthreads();
    for (int i = tid; i < CHc * 128; i += 256)
      s_w1[i] = w1[(i >> 7) * Cc + (ch << 7) + (i & 127)];
    __syncthreads();
    for (int c = 0; c < 128; ++c) {
      float xv = xp[(size_t)((ch << 7) + c) * Nc];
      #pragma unroll
      for (int o = 0; o < CHc; ++o)
        acc[o] = fmaf(s_w1[o * 128 + c], xv, acc[o]);
    }
  }
  float dot = b2[0];
  #pragma unroll
  for (int o = 0; o < CHc; ++o) {
    float hv = acc[o];
    float g = 0.5f * hv * (1.0f + erff(hv * 0.70710678118654752f));  // exact GELU
    dot = fmaf(g, w2[o], dot);
  }
  float impv = 1.0f / (1.0f + expf(-dot)) + 0.5f * bmap[(size_t)b * Nc + p];
  imp_ws[(size_t)b * Nc + p] = impv;
  imp_out[(size_t)b * Nc + p] = impv;
}

// ---------------- 2) top-K via in-LDS bitonic sort ----------------

__global__ __launch_bounds__(1024) void k_topk(const float* __restrict__ imp,
                                               int* __restrict__ idx_out)
{
  __shared__ float sv[Nc];
  __shared__ int   si[Nc];
  const int b = blockIdx.x;
  for (int i = threadIdx.x; i < Nc; i += 1024) { sv[i] = imp[b * Nc + i]; si[i] = i; }
  __syncthreads();
  for (int k = 2; k <= Nc; k <<= 1) {
    for (int j = k >> 1; j > 0; j >>= 1) {
      for (int t = threadIdx.x; t < (Nc >> 1); t += 1024) {
        int i  = ((t & ~(j - 1)) << 1) | (t & (j - 1));
        int p2 = i | j;
        bool asc = (i & k) != 0;              // overall descending sort
        float a = sv[i], c = sv[p2];
        if (asc ? (a > c) : (a < c)) {
          sv[i] = c; sv[p2] = a;
          int ti = si[i]; si[i] = si[p2]; si[p2] = ti;
        }
      }
      __syncthreads();
    }
  }
  if (threadIdx.x < Kc) idx_out[b * Kc + threadIdx.x] = si[threadIdx.x];
}

// ---------------- 3) gather selected tokens -> bf16 ----------------

__global__ __launch_bounds__(256) void k_gather(const float* __restrict__ x,
                                                const int* __restrict__ idx,
                                                bf16_t* __restrict__ xsp)
{
  size_t e = (size_t)blockIdx.x * 256 + threadIdx.x;    // over B*K*C
  int b = (int)(e / ((size_t)Kc * Cc));
  int r = (int)(e % ((size_t)Kc * Cc));
  int k = r >> 8;
  int c = r & 255;
  int pix = idx[b * Kc + k];
  xsp[e] = (bf16_t)x[((size_t)b * Cc + c) * Nc + pix];
}

// ---------------- 4) weights -> bf16 ----------------

__global__ __launch_bounds__(256) void k_cvtw(const float* __restrict__ wq,
                                              const float* __restrict__ wk,
                                              const float* __restrict__ wv,
                                              const float* __restrict__ wo,
                                              bf16_t* __restrict__ out)
{
  int i = blockIdx.x * 256 + threadIdx.x;   // 0..65535
  int m = blockIdx.y;
  const float* src = (m == 0) ? wq : (m == 1) ? wk : (m == 2) ? wv : wo;
  out[(size_t)m * 65536 + i] = (bf16_t)src[i];
}

// ---------------- 5) QKV projection GEMMs (WMMA + TDM-staged A block) ----------
// All 8 waves of a block share the same 16-token A block: stage it once via the
// Tensor Data Mover, then each wave reads its A operands with ds_load_b128.
// Q/K stored [b][t][c] bf16; V stored transposed [b][h][d][t] bf16.

__global__ __launch_bounds__(256) void k_qkv(
    const bf16_t* __restrict__ xsp, const bf16_t* __restrict__ wbf,
    const float* __restrict__ bq, const float* __restrict__ bk,
    const float* __restrict__ bv,
    bf16_t* __restrict__ qout, bf16_t* __restrict__ kout, bf16_t* __restrict__ vTout)
{
  __shared__ bf16_t s_a[16 * Cc];           // 8 KB staged A block
  const int lane = threadIdx.x & 31;
  const int wid  = threadIdx.x >> 5;
  const int which = blockIdx.y;
  const bf16_t* w = wbf + (size_t)which * 65536;
  const float* bias = (which == 0) ? bq : (which == 1) ? bk : bv;
  const int tile0 = blockIdx.x * 8;         // block-uniform
  const int b  = tile0 >> 10;
  const int tm = (tile0 & 1023) >> 4;       // uniform: 8 tiles share tm
  const int tn = (tile0 & 15) + wid;        // per-wave output-column tile
  const bf16_t* abase = xsp + (size_t)b * Kc * Cc + (size_t)tm * 16 * Cc;
#if HAVE_TDM
  if (wid == 0) tdm_stage_16x256(s_a, abase);   // issues + s_wait_tensorcnt 0
#else
  for (int i = threadIdx.x; i < 16 * Cc / 8; i += 256)
    ((uint4*)s_a)[i] = ((const uint4*)abase)[i];
#endif
  __syncthreads();

  v8f acc = {};
  for (int kk = 0; kk < Cc; kk += 32) {
    v16bf a  = load_tile_lds(s_a, kk, lane);
    v16bf bm = load_tile(w, tn * 16, Cc, kk, lane);   // W rows == B^T
    acc = wmma_bf16(a, bm, acc);
  }
  const int l15 = lane & 15, half = lane >> 4;
  const int col = tn * 16 + l15;
  const float bsv = bias[col];
  if (which < 2) {
    bf16_t* out = ((which == 0) ? qout : kout) + (size_t)b * Kc * Cc;
    const int row0 = tm * 16 + half * 8;
    #pragma unroll
    for (int g = 0; g < 8; ++g)
      out[(size_t)(row0 + g) * Cc + col] = (bf16_t)(acc[g] + bsv);
  } else {
    const int h = col >> 5, d = col & 31;
    union { bf16_t hh[8]; uint4 u; } pk;
    #pragma unroll
    for (int g = 0; g < 8; ++g) pk.hh[g] = (bf16_t)(acc[g] + bsv);
    bf16_t* out = vTout + (((size_t)b * NHc + h) * HDc + d) * Kc
                        + tm * 16 + half * 8;         // 8 consecutive tokens
    *(uint4*)out = pk.u;
  }
}

// ---------------- 6) sparse attention, flash-style online softmax ----------------

__global__ __launch_bounds__(256) void k_attn(
    const bf16_t* __restrict__ q, const bf16_t* __restrict__ kmat,
    const bf16_t* __restrict__ vT, bf16_t* __restrict__ o)
{
  __shared__ bf16_t s_p[8][16 * 32];        // per-wave P tile (D-layout -> A-layout bounce)
  const int lane = threadIdx.x & 31;
  const int w = threadIdx.x >> 5;
  const int l15 = lane & 15, half = lane >> 4;
  int wave = blockIdx.x * 8 + w;            // 0..4095
  int b = wave >> 9;
  int rem = wave & 511;
  int h = rem >> 6;
  int tq = rem & 63;
  const float scale = 0.17677669529663687f; // 1/sqrt(32)
  const bf16_t* qb = q + (size_t)b * Kc * Cc;
  const bf16_t* kb = kmat + (size_t)b * Kc * Cc;
  const bf16_t* vb = vT + ((size_t)b * NHc + h) * (size_t)HDc * Kc;
  v16bf aq = load_tile(qb, tq * 16, Cc, h * HDc, lane);
  v8f o0 = {}, o1 = {};
  float mrow[8], lrow[8];
  #pragma unroll
  for (int g = 0; g < 8; ++g) { mrow[g] = -1e30f; lrow[g] = 0.0f; }

  for (int kt = 0; kt < (Kc / 16); kt += 2) {
    v16bf kt0 = load_tile(kb, kt * 16,       Cc, h * HDc, lane);
    v16bf kt1 = load_tile(kb, (kt + 1) * 16, Cc, h * HDc, lane);
    // prefetch next key/value block into cache while we do softmax VALU work
    if (kt + 2 < (Kc / 16)) {
      __builtin_prefetch(kb + ((size_t)(kt + 2) * 16 + l15) * Cc + h * HDc, 0, 3);
      __builtin_prefetch(kb + ((size_t)(kt + 3) * 16 + l15) * Cc + h * HDc, 0, 3);
      __builtin_prefetch(vb + (size_t)l15 * Kc + (kt + 2) * 16, 0, 3);
      __builtin_prefetch(vb + (size_t)(16 + l15) * Kc + (kt + 2) * 16, 0, 3);
    }
    v8f z = {};
    v8f s0 = wmma_bf16(aq, kt0, z);          // S = Q K^T (hd=32 -> one WMMA)
    v8f s1 = wmma_bf16(aq, kt1, z);
    float p0[8], p1[8];
    #pragma unroll
    for (int g = 0; g < 8; ++g) {
      float a0 = s0[g] * scale, a1 = s1[g] * scale;
      float pm = fmaxf(a0, a1);
      #pragma unroll
      for (int off = 1; off < 16; off <<= 1) pm = fmaxf(pm, __shfl_xor(pm, off, 32));
      float m2 = fmaxf(mrow[g], pm);
      float corr = expf(mrow[g] - m2);
      mrow[g] = m2;
      p0[g] = expf(a0 - m2);
      p1[g] = expf(a1 - m2);
      float ps = p0[g] + p1[g];
      #pragma unroll
      for (int off = 1; off < 16; off <<= 1) ps += __shfl_xor(ps, off, 32);
      lrow[g] = lrow[g] * corr + ps;
      o0[g] *= corr;
      o1[g] *= corr;
    }
    // P (D-layout) -> LDS -> A-layout
    #pragma unroll
    for (int g = 0; g < 8; ++g) {
      s_p[w][(g + 8 * half) * 32 + l15]      = (bf16_t)p0[g];
      s_p[w][(g + 8 * half) * 32 + 16 + l15] = (bf16_t)p1[g];
    }
    __syncthreads();
    union { uint4 u[2]; v16bf v; } pa;
    const bf16_t* pp = &s_p[w][l15 * 32 + half * 8];
    pa.u[0] = *(const uint4*)pp;
    pa.u[1] = *(const uint4*)(pp + 16);
    v16bf vb0 = load_tile(vb, 0,  Kc, kt * 16, lane);  // V^T rows d=0..15, 32 tokens
    v16bf vb1 = load_tile(vb, 16, Kc, kt * 16, lane);  // d=16..31
    o0 = wmma_bf16(pa.v, vb0, o0);
    o1 = wmma_bf16(pa.v, vb1, o1);
    __syncthreads();
  }
  bf16_t* ob = o + (size_t)b * Kc * Cc;
  #pragma unroll
  for (int g = 0; g < 8; ++g) {
    int tok = tq * 16 + g + 8 * half;
    float inv = 1.0f / lrow[g];
    ob[(size_t)tok * Cc + h * HDc + l15]      = (bf16_t)(o0[g] * inv);
    ob[(size_t)tok * Cc + h * HDc + 16 + l15] = (bf16_t)(o1[g] * inv);
  }
}

// ---------------- 7) out-proj GEMM + residual + LayerNorm + scatter ----------------

__global__ __launch_bounds__(256) void k_out(
    const bf16_t* __restrict__ o, const bf16_t* __restrict__ wbf,
    const float* __restrict__ bo, const float* __restrict__ x,
    const int* __restrict__ idx, const float* __restrict__ ln_g,
    const float* __restrict__ ln_b, float* __restrict__ out)
{
  __shared__ float  s_y[16 * 256];          // full 16-token rows for LayerNorm
  __shared__ bf16_t s_a[16 * Cc];           // TDM-staged A block (o rows)
  const int lane = threadIdx.x & 31;
  const int w = threadIdx.x >> 5;
  const int l15 = lane & 15, half = lane >> 4;
  int b = blockIdx.x >> 6;
  int tq = blockIdx.x & 63;
  const bf16_t* ob = o + (size_t)b * Kc * Cc;
  const bf16_t* wo = wbf + (size_t)3 * 65536;
  const int* idxb = idx + b * Kc;
  const bf16_t* abase = ob + (size_t)tq * 16 * Cc;
#if HAVE_TDM
  if (w == 0) tdm_stage_16x256(s_a, abase);
#else
  for (int i = threadIdx.x; i < 16 * Cc / 8; i += 256)
    ((uint4*)s_a)[i] = ((const uint4*)abase)[i];
#endif
  __syncthreads();

  for (int t2 = 0; t2 < 2; ++t2) {
    int tn = w * 2 + t2;
    v8f acc = {};
    for (int kk = 0; kk < Cc; kk += 32) {
      v16bf a  = load_tile_lds(s_a, kk, lane);
      v16bf bm = load_tile(wo, tn * 16, Cc, kk, lane);
      acc = wmma_bf16(a, bm, acc);
    }
    int col = tn * 16 + l15;
    float bsv = bo[col];
    #pragma unroll
    for (int g = 0; g < 8; ++g) {
      int m = g + 8 * half;
      int tok = idxb[tq * 16 + m];
      float y = acc[g] + bsv + x[((size_t)b * Cc + col) * Nc + tok];  // residual (fp32 x)
      s_y[m * 256 + col] = y;
    }
  }
  __syncthreads();
  int r  = threadIdx.x >> 4;                // token row 0..15
  int c0 = threadIdx.x & 15;
  float s1 = 0.f, s2 = 0.f;
  #pragma unroll
  for (int j = 0; j < 16; ++j) {
    float v = s_y[r * 256 + c0 + 16 * j];
    s1 += v; s2 += v * v;
  }
  #pragma unroll
  for (int off = 1; off < 16; off <<= 1) {
    s1 += __shfl_xor(s1, off, 32);
    s2 += __shfl_xor(s2, off, 32);
  }
  float mu  = s1 * (1.0f / 256.0f);
  float var = s2 * (1.0f / 256.0f) - mu * mu;
  float rstd = rsqrtf(var + 1e-5f);
  int tok = idxb[tq * 16 + r];
  float* outb = out + (size_t)b * Cc * Nc + tok;
  #pragma unroll
  for (int j = 0; j < 16; ++j) {
    int c = c0 + 16 * j;
    float v2 = (s_y[r * 256 + c] - mu) * rstd * ln_g[c] + ln_b[c];
    outb[(size_t)c * Nc] = v2;
  }
}

// ---------------- launch ----------------

#define OFF_IMP ((size_t)0)
#define OFF_IDX (OFF_IMP + (size_t)Bc * Nc * 4)
#define OFF_XSP (OFF_IDX + (size_t)Bc * Kc * 4)
#define OFF_Q   (OFF_XSP + (size_t)Bc * Kc * Cc * 2)
#define OFF_K   (OFF_Q   + (size_t)Bc * Kc * Cc * 2)
#define OFF_VT  (OFF_K   + (size_t)Bc * Kc * Cc * 2)
#define OFF_O   (OFF_VT  + (size_t)Bc * Kc * Cc * 2)
#define OFF_WBF (OFF_O   + (size_t)Bc * Kc * Cc * 2)

extern "C" void kernel_launch(void* const* d_in, const int* in_sizes, int n_in,
                              void* d_out, int out_size, void* d_ws, size_t ws_size,
                              hipStream_t stream) {
  (void)in_sizes; (void)n_in; (void)out_size; (void)ws_size;
  const float* x    = (const float*)d_in[0];
  const float* bmap = (const float*)d_in[1];
  const float* w1   = (const float*)d_in[2];
  const float* b1   = (const float*)d_in[3];
  const float* w2   = (const float*)d_in[4];
  const float* b2   = (const float*)d_in[5];
  const float* wq   = (const float*)d_in[6];
  const float* bq   = (const float*)d_in[7];
  const float* wk   = (const float*)d_in[8];
  const float* bk   = (const float*)d_in[9];
  const float* wvp  = (const float*)d_in[10];
  const float* bv   = (const float*)d_in[11];
  const float* wo   = (const float*)d_in[12];
  const float* bo   = (const float*)d_in[13];
  const float* lng  = (const float*)d_in[14];
  const float* lnb  = (const float*)d_in[15];

  float* out     = (float*)d_out;
  float* imp_out = out + (size_t)Bc * Cc * Nc;

  char* ws = (char*)d_ws;
  float*  ws_imp = (float*)(ws + OFF_IMP);
  int*    ws_idx = (int*)(ws + OFF_IDX);
  bf16_t* ws_xsp = (bf16_t*)(ws + OFF_XSP);
  bf16_t* ws_q   = (bf16_t*)(ws + OFF_Q);
  bf16_t* ws_k   = (bf16_t*)(ws + OFF_K);
  bf16_t* ws_vT  = (bf16_t*)(ws + OFF_VT);
  bf16_t* ws_o   = (bf16_t*)(ws + OFF_O);
  bf16_t* ws_wbf = (bf16_t*)(ws + OFF_WBF);

  // out = x  (scatter overwrites the K selected tokens later)
  hipMemcpyAsync(d_out, d_in[0], (size_t)Bc * Cc * Nc * sizeof(float),
                 hipMemcpyDeviceToDevice, stream);

  k_importance<<<dim3(Bc * (Nc / 256)), 256, 0, stream>>>(x, bmap, w1, b1, w2, b2,
                                                          ws_imp, imp_out);
  k_topk<<<dim3(Bc), 1024, 0, stream>>>(ws_imp, ws_idx);
  k_gather<<<dim3((Bc * Kc * Cc) / 256), 256, 0, stream>>>(x, ws_idx, ws_xsp);
  k_cvtw<<<dim3(256, 4), 256, 0, stream>>>(wq, wk, wvp, wo, ws_wbf);
  k_qkv<<<dim3(1024, 3), 256, 0, stream>>>(ws_xsp, ws_wbf, bq, bk, bv,
                                           ws_q, ws_k, ws_vT);
  k_attn<<<dim3(512), 256, 0, stream>>>(ws_q, ws_k, ws_vT, ws_o);
  k_out<<<dim3(512), 256, 0, stream>>>(ws_o, ws_wbf, bo, x, ws_idx, lng, lnb, out);
}
